// STA_block_att_47021301956870
// MI455X (gfx1250) — matrix operand order repaired and checked by reading, not compiled, not run on hardware
//
#include <hip/hip_runtime.h>
#include <cstdint>

#define DEVINL __device__ __forceinline__

typedef __attribute__((ext_vector_type(16))) __bf16 v16bf;
typedef __attribute__((ext_vector_type(8)))  float  v8f;

// Problem constants (match reference)
constexpr int Bb = 16, Nn = 512, Ll = 96, IE = 64, OUTC = 96;
constexpr int TILE = IE * Ll;          // 6144 = flattened c_out row
constexpr int XSROW = IE * Nn;         // 32768 floats per (b,l) plane

DEVINL int laneid() { return threadIdx.x & 31; }
DEVINL v8f vzero8() { v8f z = {}; return z; }

DEVINL v8f wmma_bf16(v16bf a, v16bf b, v8f c) {
  // D = A(16x32 bf16) * B(32x16 bf16) + C(16x16 f32)
  return __builtin_amdgcn_wmma_f32_16x16x32_bf16(false, a, false, b, (short)0, c,
                                                 false, false);
}

// ---- CDNA5 async global->LDS copy (ASYNCcnt path) -----------------------
// Copies `bytes` (multiple of 4096 here) from global to LDS, 16B per lane.
DEVINL void async_copy_to_lds(const void* gsrc, void* ldst, int bytes, int tid) {
  for (int off = tid * 16; off < bytes; off += 256 * 16) {
    unsigned lo = (unsigned)(size_t)((char*)ldst + off);          // LDS byte offset
    unsigned long long ga = (unsigned long long)((const char*)gsrc + off);
    asm volatile("global_load_async_to_lds_b128 %0, %1, off"
                 :: "v"(lo), "v"(ga) : "memory");
  }
}
DEVINL void wait_async0() {
  asm volatile("s_wait_asynccnt 0x0" ::: "memory");
}

// ---- fragment loaders (ISA 7.12.2 layouts, wave32) ----------------------
// A fragment from LDS row-major [M][K] (pitch mult of 8 elems)
DEVINL v16bf frag_a_lds(const __bf16* base, int pitch, int mb, int kb) {
  int lane = laneid();
  const __bf16* p = base + (mb + (lane & 15)) * pitch + kb + ((lane >> 4) << 3);
  v16bf v;
  ((uint4*)&v)[0] = *(const uint4*)(p);
  ((uint4*)&v)[1] = *(const uint4*)(p + 16);
  return v;
}
// B fragment from LDS holding B^T row-major [N][K]
DEVINL v16bf frag_b_lds(const __bf16* base, int pitch, int nb, int kb) {
  int lane = laneid();
  const __bf16* p = base + (nb + (lane & 15)) * pitch + kb + ((lane >> 4) << 3);
  v16bf v;
  ((uint4*)&v)[0] = *(const uint4*)(p);
  ((uint4*)&v)[1] = *(const uint4*)(p + 16);
  return v;
}
// A fragment from global f32 row-major [M][K]
DEVINL v16bf frag_a_g32(const float* base, int ld, int mb, int kb) {
  int lane = laneid();
  const float* p = base + (size_t)(mb + (lane & 15)) * ld + kb + ((lane >> 4) << 3);
  v16bf v;
#pragma unroll
  for (int j = 0; j < 8; ++j) v[j] = (__bf16)p[j];
#pragma unroll
  for (int j = 0; j < 8; ++j) v[8 + j] = (__bf16)p[16 + j];
  return v;
}
// B fragment from global f32 B^T row-major [N][K]; optional next-K prefetch
template <bool PF>
DEVINL v16bf frag_b_g32_t(const float* base, int ld, int nb, int kb) {
  int lane = laneid();
  const float* p = base + (size_t)(nb + (lane & 15)) * ld + kb + ((lane >> 4) << 3);
  if (PF) __builtin_prefetch(p + 32, 0, 3);  // next k-step of this row -> global_prefetch
  v16bf v;
#pragma unroll
  for (int j = 0; j < 16; ++j) v[j] = (__bf16)p[(j & 7) + ((j >> 3) << 4)];
  return v;
}
DEVINL v16bf frag_b_g32(const float* base, int ld, int nb, int kb) {
  return frag_b_g32_t<false>(base, ld, nb, kb);
}
// D fragment -> LDS row-major bf16
DEVINL void store_d_rm(__bf16* base, int pitch, int mb, int nb, const v8f& a) {
  int lane = laneid();
  int col = nb + (lane & 15);
  int row0 = mb + ((lane >> 4) << 3);
#pragma unroll
  for (int r = 0; r < 8; ++r) base[(row0 + r) * pitch + col] = (__bf16)a[r];
}
// D fragment -> LDS transposed (row = N, 8 contiguous M cols -> one b128)
DEVINL void store_d_tr(__bf16* base, int pitch, int mb, int nb, const v8f& a) {
  int lane = laneid();
  int row = nb + (lane & 15);
  int col0 = mb + ((lane >> 4) << 3);
  alignas(16) __bf16 t[8];
#pragma unroll
  for (int r = 0; r < 8; ++r) t[r] = (__bf16)a[r];
  *(uint4*)(base + row * pitch + col0) = *(const uint4*)t;
}

DEVINL float sum8(const float* p) {
  float s = 0.f;
#pragma unroll
  for (int i = 0; i < 8; ++i) s += p[i];
  return s;
}

// row-wise softmax of Sb[64][64] (pitch 72) -> P[64][72]; 256 threads, 4/row
DEVINL void softmax64(const __bf16* Sb, __bf16* P) {
  int row = threadIdx.x >> 2, q4 = threadIdx.x & 3;
  float e[16];
  float mx = -3.0e38f;
#pragma unroll
  for (int j = 0; j < 16; ++j) {
    e[j] = (float)Sb[row * 72 + q4 * 16 + j];
    mx = fmaxf(mx, e[j]);
  }
  mx = fmaxf(mx, __shfl_xor(mx, 1));
  mx = fmaxf(mx, __shfl_xor(mx, 2));
  float s = 0.f;
#pragma unroll
  for (int j = 0; j < 16; ++j) { e[j] = __expf(e[j] - mx); s += e[j]; }
  s += __shfl_xor(s, 1);
  s += __shfl_xor(s, 2);
  float inv = 1.0f / s;
#pragma unroll
  for (int j = 0; j < 16; ++j) P[row * 72 + q4 * 16 + j] = (__bf16)(e[j] * inv);
}

// bf16-pack conversion pass: LDS f32 [96][64] -> LDS bf16 [96][72]
DEVINL void cvt_tile_to_bf16(const float* Xf, __bf16* Xt, int tid) {
  for (int idx = tid; idx < Ll * 32; idx += 256) {
    int l = idx >> 5, c0 = (idx & 31) * 2;
    float2 f = *(const float2*)(Xf + l * IE + c0);
    union { unsigned u; __bf16 h[2]; } pk;
    pk.h[0] = (__bf16)f.x; pk.h[1] = (__bf16)f.y;
    *(unsigned*)(Xt + l * 72 + c0) = pk.u;
  }
}

// =========================================================================
// Kernel 1: time attention. One workgroup per (b,n). Output T in [B,N,L,IE].
// dyn LDS = 54272 B
__global__ __launch_bounds__(256) void time_att_kernel(
    const float* __restrict__ x,
    const float* __restrict__ wq, const float* __restrict__ bq,
    const float* __restrict__ wk, const float* __restrict__ bk,
    const float* __restrict__ wvm, const float* __restrict__ bvm,
    const float* __restrict__ lnw, const float* __restrict__ lnb,
    const float* __restrict__ hw, const float* __restrict__ hb,
    float* __restrict__ T) {
  extern __shared__ char smem[];
  __bf16* Xt = (__bf16*)smem;             // [96][72] bf16 (phase A: B^T for QKV)
  __bf16* Sb = (__bf16*)smem;             // [64][72] (phase B/C overlay)
  __bf16* Z  = (__bf16*)smem;             // [64][104] (phase D/E overlay)
  float*  st = (float*)(smem + 13312);    // mean[64], rstd[64]
  float*  Xf = (float*)(smem + 13824);    // [96][64] f32 async staging (overlay Q/K)
  __bf16* Q  = (__bf16*)(smem + 13824);   // [64][104]
  __bf16* P  = (__bf16*)(smem + 13824);   // [64][72] overlay after B
  __bf16* K  = (__bf16*)(smem + 27136);   // [64][104]
  __bf16* Vt = (__bf16*)(smem + 40448);   // [96][72] (V^T: [l][o])

  const int n = blockIdx.x, b = blockIdx.y;
  const int tid = threadIdx.x, wid = tid >> 5, lane = tid & 31;
  const float* xt = x + (size_t)(b * Nn + n) * TILE;  // [l][ie] row-major
  const float tsc = sum8(hw), tbias = hb[0];

  // Phase A: async-stage x tile (24 KB) -> Xf (LDS f32), then pack to bf16
  async_copy_to_lds(xt, Xf, TILE * 4, tid);
  wait_async0();
  __syncthreads();
  cvt_tile_to_bf16(Xf, Xt, tid);
  __syncthreads();

  // QKV: 3 x (4x6) tiles = 72, 9 per wave. A = W (global f32), B^T = Xt.
  for (int t = 0; t < 9; ++t) {
    int id = wid * 9 + t;
    int mat = id / 24, r24 = id % 24;
    int mb = (r24 & 3) * 16, nb = (r24 >> 2) * 16;
    const float* W = mat == 0 ? wq : (mat == 1 ? wk : wvm);
    const float* bias = mat == 0 ? bq : (mat == 1 ? bk : bvm);
    v8f acc = vzero8();
    for (int kb = 0; kb < IE; kb += 32)
      acc = wmma_bf16(frag_a_g32(W, IE, mb, kb), frag_b_lds(Xt, 72, nb, kb), acc);
    int row0 = mb + ((lane >> 4) << 3);
#pragma unroll
    for (int r = 0; r < 8; ++r) acc[r] += bias[row0 + r];
    if (mat == 2) store_d_tr(Vt, 72, mb, nb, acc);
    else          store_d_rm(mat == 0 ? Q : K, 104, mb, nb, acc);
  }
  __syncthreads();

  // Phase B: S = (Q K^T) / sqrt(IE/H); 16 tiles, 2 per wave
  {
    const float rs = 0.35355339059327373f;  // 1/sqrt(8)
    int mb = (wid & 3) * 16;
#pragma unroll
    for (int t = 0; t < 2; ++t) {
      int nb = ((wid >> 2) * 2 + t) * 16;
      v8f acc = vzero8();
      for (int kb = 0; kb < Ll; kb += 32)
        acc = wmma_bf16(frag_a_lds(Q, 104, mb, kb), frag_b_lds(K, 104, nb, kb), acc);
#pragma unroll
      for (int r = 0; r < 8; ++r) acc[r] *= rs;
      store_d_rm(Sb, 72, mb, nb, acc);
    }
  }
  __syncthreads();
  softmax64(Sb, P);
  __syncthreads();

  // Phase D: line = P @ V; +x residual, scale, -> Z[ie][l]
  {
    int mb = (wid & 3) * 16;
#pragma unroll
    for (int t = 0; t < 3; ++t) {
      int nb = ((wid >> 2) * 3 + t) * 16;
      v8f acc = vzero8();
      for (int kb = 0; kb < IE; kb += 32)
        acc = wmma_bf16(frag_a_lds(P, 72, mb, kb), frag_b_lds(Vt, 72, nb, kb), acc);
      int l = nb + (lane & 15);
      int ie0 = mb + ((lane >> 4) << 3);
      const float* xr = xt + l * IE + ie0;
#pragma unroll
      for (int r = 0; r < 8; ++r)
        Z[(ie0 + r) * 104 + l] = (__bf16)(xr[r] + acc[r] * tsc + tbias);
    }
  }
  __syncthreads();

  // Phase E: LayerNorm over l (96) per ie row; write T[b][n][l][ie]
  {
    int row = tid >> 2, q4 = tid & 3;
    float s = 0.f, ss = 0.f;
#pragma unroll
    for (int i = 0; i < 24; ++i) {
      float z = (float)Z[row * 104 + q4 + i * 4];
      s += z; ss += z * z;
    }
    s += __shfl_xor(s, 1);  s += __shfl_xor(s, 2);
    ss += __shfl_xor(ss, 1); ss += __shfl_xor(ss, 2);
    if (q4 == 0) {
      float m = s * (1.0f / 96.0f);
      float v = ss * (1.0f / 96.0f) - m * m;
      st[row] = m;
      st[64 + row] = rsqrtf(v + 1e-5f);
    }
  }
  __syncthreads();
  {
    float* To = T + (size_t)(b * Nn + n) * TILE;
    for (int idx = tid; idx < TILE; idx += 256) {
      int ie = idx & 63, l = idx >> 6;
      float z = (float)Z[ie * 104 + l];
      To[idx] = (z - st[ie]) * st[64 + ie] * lnw[l] + lnb[l];
    }
  }
}

// =========================================================================
// Kernel 2: space attention. One workgroup per (b,l). Output [B,L,IE,N].
// dyn LDS = 138240 B
__global__ __launch_bounds__(256) void space_att_kernel(
    const float* __restrict__ x,
    const float* __restrict__ wq, const float* __restrict__ bq,
    const float* __restrict__ wk, const float* __restrict__ bk,
    const float* __restrict__ wvm, const float* __restrict__ bvm,
    const float* __restrict__ hw, const float* __restrict__ hb,
    float* __restrict__ SN) {
  extern __shared__ char smem[];
  __bf16* Xs  = (__bf16*)smem;              // [64][520]
  __bf16* Qt  = (__bf16*)(smem + 66560);    // [64][136]
  __bf16* Kt  = (__bf16*)(smem + 83968);    // [64][136]
  __bf16* Vtt = (__bf16*)(smem + 101376);   // [128][72]  V^T tile
  __bf16* Sb  = (__bf16*)(smem + 119808);   // [64][72]
  __bf16* P   = (__bf16*)(smem + 129024);   // [64][72]

  const int l = blockIdx.x, b = blockIdx.y;
  const int tid = threadIdx.x, wid = tid >> 5, lane = tid & 31;
  const float* xb = x + (size_t)b * Nn * TILE + (size_t)l * IE;  // (m,c) at m*6144+c
  const float ssc = sum8(hw), sbias = hb[0];

  // Stage Xs[c][m] bf16 (A-matrix, row-major over m)
  for (int idx = tid; idx < Nn * 32; idx += 256) {
    int m = idx >> 5, c0 = (idx & 31) * 2;
    float2 f = *(const float2*)(xb + (size_t)m * TILE + c0);
    Xs[c0 * 520 + m] = (__bf16)f.x;
    Xs[(c0 + 1) * 520 + m] = (__bf16)f.y;
  }
  __syncthreads();

  v8f s_acc[2];
  s_acc[0] = vzero8(); s_acc[1] = vzero8();

  // Loop 1: per 128-col n-tile, compute Q,K panels and accumulate S
  for (int nt = 0; nt < 4; ++nt) {
    int ng0 = nt * 128 + wid * 16;  // this wave's global weight-row block
#pragma unroll
    for (int mat = 0; mat < 2; ++mat) {
      const float* W = mat ? wk : wq;
      const float* bias = mat ? bk : bq;
      __bf16* dst = mat ? Kt : Qt;
      v8f acc[4];
#pragma unroll
      for (int m4 = 0; m4 < 4; ++m4) acc[m4] = vzero8();
      for (int kb = 0; kb < Nn; kb += 32) {
        v16bf bf = frag_b_g32_t<true>(W, Nn, ng0, kb);   // + global_prefetch next K
#pragma unroll
        for (int m4 = 0; m4 < 4; ++m4)
          acc[m4] = wmma_bf16(frag_a_lds(Xs, 520, m4 * 16, kb), bf, acc[m4]);
      }
      float bn = bias[ng0 + (lane & 15)];
#pragma unroll
      for (int m4 = 0; m4 < 4; ++m4) {
#pragma unroll
        for (int r = 0; r < 8; ++r) acc[m4][r] += bn;
        store_d_rm(dst, 136, m4 * 16, wid * 16, acc[m4]);
      }
    }
    __syncthreads();
    {
      int mb = (wid & 3) * 16;
#pragma unroll
      for (int t = 0; t < 2; ++t) {
        int nb = ((wid >> 2) * 2 + t) * 16;
        for (int kb = 0; kb < 128; kb += 32)
          s_acc[t] = wmma_bf16(frag_a_lds(Qt, 136, mb, kb),
                               frag_b_lds(Kt, 136, nb, kb), s_acc[t]);
      }
    }
    __syncthreads();
  }

  // S / sqrt(N/H) -> softmax
  {
    const float rs = 0.125f;  // 1/sqrt(64)
    int mb = (wid & 3) * 16;
#pragma unroll
    for (int t = 0; t < 2; ++t) {
      int nb = ((wid >> 2) * 2 + t) * 16;
      v8f a = s_acc[t];
#pragma unroll
      for (int r = 0; r < 8; ++r) a[r] *= rs;
      store_d_rm(Sb, 72, mb, nb, a);
    }
  }
  __syncthreads();
  softmax64(Sb, P);
  __syncthreads();

  // Loop 2: per n-tile, compute V panel (transposed) then line = P @ V
  float* out = SN + (size_t)(b * Ll + l) * XSROW;
  for (int nt = 0; nt < 4; ++nt) {
    int ng0 = nt * 128 + wid * 16;
    {
      v8f acc[4];
#pragma unroll
      for (int m4 = 0; m4 < 4; ++m4) acc[m4] = vzero8();
      for (int kb = 0; kb < Nn; kb += 32) {
        v16bf bf = frag_b_g32_t<true>(wvm, Nn, ng0, kb);
#pragma unroll
        for (int m4 = 0; m4 < 4; ++m4)
          acc[m4] = wmma_bf16(frag_a_lds(Xs, 520, m4 * 16, kb), bf, acc[m4]);
      }
      float bn = bvm[ng0 + (lane & 15)];
#pragma unroll
      for (int m4 = 0; m4 < 4; ++m4) {
#pragma unroll
        for (int r = 0; r < 8; ++r) acc[m4][r] += bn;
        store_d_tr(Vtt, 72, m4 * 16, wid * 16, acc[m4]);
      }
    }
    __syncthreads();
    {
      v8f lacc[4];
#pragma unroll
      for (int m4 = 0; m4 < 4; ++m4) lacc[m4] = vzero8();
      for (int kb = 0; kb < IE; kb += 32) {
        v16bf bf = frag_b_lds(Vtt, 72, wid * 16, kb);
#pragma unroll
        for (int m4 = 0; m4 < 4; ++m4)
          lacc[m4] = wmma_bf16(frag_a_lds(P, 72, m4 * 16, kb), bf, lacc[m4]);
      }
      int nglob = nt * 128 + wid * 16 + (lane & 15);
#pragma unroll
      for (int m4 = 0; m4 < 4; ++m4) {
        int c0 = m4 * 16 + ((lane >> 4) << 3);
#pragma unroll
        for (int r = 0; r < 8; ++r)
          out[(size_t)(c0 + r) * Nn + nglob] = lacc[m4][r] * ssc + sbias;
      }
    }
    __syncthreads();
  }
}

// =========================================================================
// Kernel 3: transpose [B,L,IE,N] -> [B,N,L,IE] (coalesced both sides)
__global__ __launch_bounds__(256) void transpose_kernel(
    const float* __restrict__ src, float* __restrict__ dst) {
  __shared__ float tile[32][33];
  const int bl = blockIdx.z;
  const int n0 = blockIdx.x * 32, c0 = blockIdx.y * 32;
  const int tx = threadIdx.x, ty = threadIdx.y;
  const float* s = src + (size_t)bl * XSROW;
  for (int i = ty; i < 32; i += 8) tile[i][tx] = s[(size_t)(c0 + i) * Nn + n0 + tx];
  __syncthreads();
  const int b_ = bl / Ll, l_ = bl % Ll;
  float* d = dst + (size_t)b_ * Nn * TILE + (size_t)l_ * IE;
  for (int i = ty; i < 32; i += 8)
    d[(size_t)(n0 + i) * TILE + c0 + tx] = tile[tx][i];
}

// =========================================================================
// Kernel 4: cross attention. q from Ssw ([B,N,L,IE]), k/v + residual from T.
// Output c_out rows [bn][IE*96+l]. dyn LDS = 68096 B
__global__ __launch_bounds__(256) void cross_att_kernel(
    const float* __restrict__ Ssw, const float* __restrict__ T,
    const float* __restrict__ wq, const float* __restrict__ bq,
    const float* __restrict__ wk, const float* __restrict__ bk,
    const float* __restrict__ wvm, const float* __restrict__ bvm,
    const float* __restrict__ lnw, const float* __restrict__ lnb,
    const float* __restrict__ hw, const float* __restrict__ hb,
    float* __restrict__ C) {
  extern __shared__ char smem[];
  __bf16* SQ = (__bf16*)smem;             // [96][72] (q-source B^T)
  __bf16* Sb = (__bf16*)smem;             // [64][72] overlay
  __bf16* TT = (__bf16*)(smem + 13824);   // [96][72] (k/v-source B^T)
  __bf16* Z  = (__bf16*)(smem + 13824);   // [64][104] overlay
  float*  st = (float*)(smem + 27136);    // mean[64], rstd[64]
  float*  Xf = (float*)(smem + 27648);    // [96][64] f32 async staging (overlay Q/K)
  __bf16* Q  = (__bf16*)(smem + 27648);   // [64][104]
  __bf16* P  = (__bf16*)(smem + 27648);   // [64][72] overlay
  __bf16* K  = (__bf16*)(smem + 40960);   // [64][104]
  __bf16* Vt = (__bf16*)(smem + 54272);   // [96][72]

  const int n = blockIdx.x, b = blockIdx.y;
  const int tid = threadIdx.x, wid = tid >> 5, lane = tid & 31;
  const size_t bn = (size_t)(b * Nn + n);
  const float* sq_t = Ssw + bn * TILE;
  const float* t_t  = T + bn * TILE;
  const float csc = sum8(hw), cbias = hb[0];

  // Async-stage + pack both source tiles (sequentially reusing Xf)
  async_copy_to_lds(sq_t, Xf, TILE * 4, tid);
  wait_async0();
  __syncthreads();
  cvt_tile_to_bf16(Xf, SQ, tid);
  __syncthreads();
  async_copy_to_lds(t_t, Xf, TILE * 4, tid);
  wait_async0();
  __syncthreads();
  cvt_tile_to_bf16(Xf, TT, tid);
  __syncthreads();

  for (int t = 0; t < 9; ++t) {
    int id = wid * 9 + t;
    int mat = id / 24, r24 = id % 24;
    int mb = (r24 & 3) * 16, nb = (r24 >> 2) * 16;
    const float* W = mat == 0 ? wq : (mat == 1 ? wk : wvm);
    const float* bias = mat == 0 ? bq : (mat == 1 ? bk : bvm);
    const __bf16* Bsrc = (mat == 0) ? SQ : TT;
    v8f acc = vzero8();
    for (int kb = 0; kb < IE; kb += 32)
      acc = wmma_bf16(frag_a_g32(W, IE, mb, kb), frag_b_lds(Bsrc, 72, nb, kb), acc);
    int row0 = mb + ((lane >> 4) << 3);
#pragma unroll
    for (int r = 0; r < 8; ++r) acc[r] += bias[row0 + r];
    if (mat == 2) store_d_tr(Vt, 72, mb, nb, acc);
    else          store_d_rm(mat == 0 ? Q : K, 104, mb, nb, acc);
  }
  __syncthreads();

  {
    const float rs = 0.35355339059327373f;
    int mb = (wid & 3) * 16;
#pragma unroll
    for (int t = 0; t < 2; ++t) {
      int nb = ((wid >> 2) * 2 + t) * 16;
      v8f acc = vzero8();
      for (int kb = 0; kb < Ll; kb += 32)
        acc = wmma_bf16(frag_a_lds(Q, 104, mb, kb), frag_b_lds(K, 104, nb, kb), acc);
#pragma unroll
      for (int r = 0; r < 8; ++r) acc[r] *= rs;
      store_d_rm(Sb, 72, mb, nb, acc);
    }
  }
  __syncthreads();
  softmax64(Sb, P);
  __syncthreads();

  {
    int mb = (wid & 3) * 16;
#pragma unroll
    for (int t = 0; t < 3; ++t) {
      int nb = ((wid >> 2) * 3 + t) * 16;
      v8f acc = vzero8();
      for (int kb = 0; kb < IE; kb += 32)
        acc = wmma_bf16(frag_a_lds(P, 72, mb, kb), frag_b_lds(Vt, 72, nb, kb), acc);
      int l = nb + (lane & 15);
      int ie0 = mb + ((lane >> 4) << 3);
      const float* tr = t_t + l * IE + ie0;
#pragma unroll
      for (int r = 0; r < 8; ++r)
        Z[(ie0 + r) * 104 + l] = (__bf16)(tr[r] + acc[r] * csc + cbias);
    }
  }
  __syncthreads();

  {
    int row = tid >> 2, q4 = tid & 3;
    float s = 0.f, ss = 0.f;
#pragma unroll
    for (int i = 0; i < 24; ++i) {
      float z = (float)Z[row * 104 + q4 + i * 4];
      s += z; ss += z * z;
    }
    s += __shfl_xor(s, 1);  s += __shfl_xor(s, 2);
    ss += __shfl_xor(ss, 1); ss += __shfl_xor(ss, 2);
    if (q4 == 0) {
      float m = s * (1.0f / 96.0f);
      float v = ss * (1.0f / 96.0f) - m * m;
      st[row] = m;
      st[64 + row] = rsqrtf(v + 1e-5f);
    }
  }
  __syncthreads();
  {
    float* Co = C + bn * TILE;
    for (int idx = tid; idx < TILE; idx += 256) {
      int ie = idx / Ll, l = idx - ie * Ll;
      float z = (float)Z[ie * 104 + l];
      Co[idx] = (z - st[ie]) * st[64 + ie] * lnw[l] + lnb[l];
    }
  }
}

// =========================================================================
// Kernel 5: final GEMM: out[8192,96] = C[8192,6144] @ W^T + bias
__global__ __launch_bounds__(256) void final_gemm_kernel(
    const float* __restrict__ C, const float* __restrict__ W,
    const float* __restrict__ bias, float* __restrict__ out) {
  const int wid = threadIdx.x >> 5, lane = threadIdx.x & 31;
  const int row0 = blockIdx.x * 128 + wid * 16;
  v8f acc[6];
#pragma unroll
  for (int j = 0; j < 6; ++j) acc[j] = vzero8();
  const float* Arow = C + (size_t)row0 * TILE;
  for (int kb = 0; kb < TILE; kb += 32) {
    v16bf a = frag_a_g32(Arow, TILE, 0, kb);
#pragma unroll
    for (int j = 0; j < 6; ++j)
      acc[j] = wmma_bf16(a, frag_b_g32_t<true>(W, TILE, j * 16, kb), acc[j]);
  }
#pragma unroll
  for (int j = 0; j < 6; ++j) {
    int o = j * 16 + (lane & 15);
    float bb = bias[o];
    int r0 = (lane >> 4) << 3;
#pragma unroll
    for (int r = 0; r < 8; ++r)
      out[(size_t)(row0 + r0 + r) * OUTC + o] = acc[j][r] + bb;
  }
}

// =========================================================================
extern "C" void kernel_launch(void* const* d_in, const int* in_sizes, int n_in,
                              void* d_out, int out_size, void* d_ws, size_t ws_size,
                              hipStream_t stream) {
  const float* x      = (const float*)d_in[0];
  const float* tq_w   = (const float*)d_in[1];
  const float* tq_b   = (const float*)d_in[2];
  const float* tk_w   = (const float*)d_in[3];
  const float* tk_b   = (const float*)d_in[4];
  const float* tv_w   = (const float*)d_in[5];
  const float* tv_b   = (const float*)d_in[6];
  const float* t_ln_w = (const float*)d_in[7];
  const float* t_ln_b = (const float*)d_in[8];
  const float* t_hw   = (const float*)d_in[9];
  const float* t_hb   = (const float*)d_in[10];
  const float* sq_w   = (const float*)d_in[11];
  const float* sq_b   = (const float*)d_in[12];
  const float* sk_w   = (const float*)d_in[13];
  const float* sk_b   = (const float*)d_in[14];
  const float* sv_w   = (const float*)d_in[15];
  const float* sv_b   = (const float*)d_in[16];
  const float* s_hw   = (const float*)d_in[17];
  const float* s_hb   = (const float*)d_in[18];
  const float* cq_w   = (const float*)d_in[19];
  const float* cq_b   = (const float*)d_in[20];
  const float* ck_w   = (const float*)d_in[21];
  const float* ck_b   = (const float*)d_in[22];
  const float* cv_w   = (const float*)d_in[23];
  const float* cv_b   = (const float*)d_in[24];
  const float* c_ln_w = (const float*)d_in[25];
  const float* c_ln_b = (const float*)d_in[26];
  const float* c_hw   = (const float*)d_in[27];
  const float* c_hb   = (const float*)d_in[28];
  const float* out_w  = (const float*)d_in[29];
  const float* out_b  = (const float*)d_in[30];
  (void)in_sizes; (void)n_in; (void)out_size; (void)ws_size;

  // Workspace layout (floats): T | SN | SS ; C aliases SN (dead after K3).
  constexpr size_t PLANE = (size_t)Bb * Nn * TILE;  // 50,331,648 floats each
  float* T  = (float*)d_ws;
  float* SN = T + PLANE;
  float* SS = SN + PLANE;
  float* C  = SN;
  float* out = (float*)d_out;

  // Allow >64KB dynamic LDS (no-op in compile-only; required on real HW)
  (void)hipFuncSetAttribute((const void*)space_att_kernel,
                            hipFuncAttributeMaxDynamicSharedMemorySize, 138240);
  (void)hipFuncSetAttribute((const void*)cross_att_kernel,
                            hipFuncAttributeMaxDynamicSharedMemorySize, 68096);
  (void)hipFuncSetAttribute((const void*)time_att_kernel,
                            hipFuncAttributeMaxDynamicSharedMemorySize, 54272);

  time_att_kernel<<<dim3(Nn, Bb), 256, 54272, stream>>>(
      x, tq_w, tq_b, tk_w, tk_b, tv_w, tv_b, t_ln_w, t_ln_b, t_hw, t_hb, T);
  space_att_kernel<<<dim3(Ll, Bb), 256, 138240, stream>>>(
      x, sq_w, sq_b, sk_w, sk_b, sv_w, sv_b, s_hw, s_hb, SN);
  transpose_kernel<<<dim3(Nn / 32, IE / 32, Bb * Ll), dim3(32, 8), 0, stream>>>(SN, SS);
  cross_att_kernel<<<dim3(Nn, Bb), 256, 68096, stream>>>(
      SS, T, cq_w, cq_b, ck_w, ck_b, cv_w, cv_b, c_ln_w, c_ln_b, c_hw, c_hb, C);
  final_gemm_kernel<<<64, 256, 0, stream>>>(C, out_w, out_b, out);
}
